// GraphConv2d_41308995453321
// MI455X (gfx1250) — matrix operand order repaired and checked by reading, not compiled, not run on hardware
//
#include <hip/hip_runtime.h>

#define BB 8
#define CC 192
#define NN 3136
#define KE 9
#define CO 384
#define TWO_C 384

typedef __attribute__((ext_vector_type(16))) _Float16 v16h;
typedef __attribute__((ext_vector_type(8)))  float    v8f;
typedef __attribute__((ext_vector_type(4)))  float    v4f;
typedef __attribute__((ext_vector_type(4)))  int      v4i;

#if __has_builtin(__builtin_amdgcn_global_load_async_to_lds_b128)
#define USE_ASYNC_LDS 1
#define GLOBAL_AS __attribute__((address_space(1)))
#define LDS_AS __attribute__((address_space(3)))
#endif

// ---------------- Prep 1: x [B,C,N] f32 -> xT [B,N,C] f16 ----------------
__global__ __launch_bounds__(256) void xt_kernel(const float* __restrict__ x,
                                                 _Float16* __restrict__ xT) {
  int idx = blockIdx.x * 256 + threadIdx.x;
  if (idx >= BB * CC * NN) return;
  int n = idx % NN;
  int c = (idx / NN) % CC;
  int b = idx / (NN * CC);
  xT[((size_t)b * NN + n) * CC + c] = (_Float16)x[idx];
}

// ---- Prep 2: Wcat = [W1-W2 | W2] pre-swizzled into A-fragment layout ----
// Afrag[mtile 0..23][ktile 0..11][lane 0..31][slot 0..15] f16
// A 16x32 k order used here (contiguous, B matches): lane<16 slots -> K {0..15}; lane>=16 -> {16..31}
__global__ __launch_bounds__(256) void wprep_kernel(const float* __restrict__ W,
                                                    _Float16* __restrict__ Af) {
  int t = blockIdx.x * 256 + threadIdx.x;  // 24*12*32*16 = 147456
  if (t >= 24 * 12 * 32 * 16) return;
  int slot = t & 15;
  int lane = (t >> 4) & 31;
  int kt   = (t >> 9) % 12;
  int mt   = (t >> 9) / 12;
  int m  = mt * 16 + (lane & 15);
  int kk = ((lane >> 4) << 4) + slot;
  int k  = kt * 32 + kk;                       // 0..383 contraction index
  float wv = (k < CC) ? (W[m * TWO_C + k] - W[m * TWO_C + k + CC])
                      : W[m * TWO_C + k];
  Af[t] = (_Float16)wv;
}

// ---------------- Main: per-batch GEMM + bias + relu + k-max ----------------
// grid: x=196 colblocks (16 nodes / 144 edge cols), y=batch.
// 384 threads = 12 waves; wave w owns row tiles {2w, 2w+1} -> all 384 rows.
__global__ __launch_bounds__(384) void edgeconv_kernel(
    const _Float16* __restrict__ xT,      // [B][N][C]
    const _Float16* __restrict__ Afrag,   // [24][12][32][16]
    const long long* __restrict__ eidx,   // [2][B][N][K]
    const float* __restrict__ bias,       // [384]
    float* __restrict__ out) {            // [B][CO][N] (pre-zeroed)
  const int colblock = blockIdx.x;
  const int b        = blockIdx.y;
  const int t        = threadIdx.x;
  const int wave     = t >> 5;             // 0..11
  const int lane     = t & 31;

  __shared__ _Float16 Bsh[144 * 200];  // one K-half: [col][192 + 8 pad] = 57.6 KB
  __shared__ int Ish[144];

  const int ebase = colblock * 144;    // edges [ebase, ebase+144) = 16 whole nodes
  v8f acc[2][9];
#pragma unroll
  for (int rt = 0; rt < 2; rt++)
#pragma unroll
    for (int i = 0; i < 9; i++) acc[rt][i] = (v8f){0.f,0.f,0.f,0.f,0.f,0.f,0.f,0.f};

  const _Float16* xTb = xT + (size_t)b * NN * CC;
  const int colr  = lane & 15;
  const int khalf = (lane >> 4) << 4;      // B-frag K sub-offset

  // A-fragment base for this wave's two row tiles: stride between ktiles = 512 f16.
  const _Float16* aw = Afrag + ((size_t)(wave * 2) * 12 * 32 + lane) * 16;

  for (int half = 0; half < 2; half++) {
    if (half) __syncthreads();  // previous WMMA phase done before Bsh overwrite
    // half 0 -> x_i uses edge_index[1]; half 1 -> x_j uses edge_index[0]
    if (t < 144) {
      int e = ebase + t;
      int n = e / 9, k = e - n * 9;
      size_t off = (size_t)(half ? 0 : 1) * ((size_t)BB * NN * KE) +
                   ((size_t)b * NN + n) * KE + k;
      Ish[t] = (int)eidx[off];
    }
    __syncthreads();

    // Gather: 144 cols x 192 f16, in 16-byte chunks (24 per column)
    for (int q = t; q < 144 * 24; q += 384) {
      int col = q / 24, c8 = q - col * 24;
      int node = Ish[col];
#ifdef USE_ASYNC_LDS
      __builtin_amdgcn_global_load_async_to_lds_b128(
          (GLOBAL_AS v4i*)(xTb + (size_t)node * CC) + c8,
          (LDS_AS v4i*)(Bsh + col * 200) + c8, 0, 0);
#else
      v4f v = *((const v4f*)(xTb + (size_t)node * CC) + c8);
      *((v4f*)(Bsh + col * 200) + c8) = v;
#endif
    }
#ifdef USE_ASYNC_LDS
    asm volatile("s_wait_asynccnt 0x0" ::: "memory");
#endif
    __syncthreads();

    // GEMM over this K-half: 6 ktiles x 9 col tiles x 2 row tiles per wave.
    // B fragments double-buffered so the next ds_load pair is in flight
    // while the current pair of WMMAs executes.
    for (int kt = 0; kt < 6; kt++) {
      int ktg = half * 6 + kt;
      v16h a0 = *(const v16h*)(aw + (size_t)ktg * 512);
      v16h a1 = *(const v16h*)(aw + (size_t)(12 + ktg) * 512);
      int koff = kt * 32 + khalf;

      union { v16h h; v4f f[2]; } bf[2];
      {
        const v4f* bp = (const v4f*)(Bsh + colr * 200 + koff);
        bf[0].f[0] = bp[0];
        bf[0].f[1] = bp[1];
      }
#pragma unroll
      for (int ct = 0; ct < 9; ct++) {
        if (ct < 8) {
          const v4f* bp = (const v4f*)(Bsh + ((ct + 1) * 16 + colr) * 200 + koff);
          bf[(ct + 1) & 1].f[0] = bp[0];
          bf[(ct + 1) & 1].f[1] = bp[1];
        }
        acc[0][ct] = __builtin_amdgcn_wmma_f32_16x16x32_f16(
            false, a0, false, bf[ct & 1].h, (short)0, acc[0][ct], false, false);
        acc[1][ct] = __builtin_amdgcn_wmma_f32_16x16x32_f16(
            false, a1, false, bf[ct & 1].h, (short)0, acc[1][ct], false, false);
      }
    }
  }

  // Epilogue: bias + relu + atomic max over the 9 edges of each node.
  // C/D layout: VGPR r -> M = r (lanes 0-15) or r+8 (lanes 16-31); N = lane%16.
#pragma unroll
  for (int rt = 0; rt < 2; rt++) {
    const int obase = (wave * 2 + rt) * 16 + ((lane >> 4) << 3);
#pragma unroll
    for (int ct = 0; ct < 9; ct++) {
      int e = ebase + ct * 16 + colr;
      int n = e / 9;
#pragma unroll
      for (int r = 0; r < 8; r++) {
        int o = obase + r;
        float v = acc[rt][ct][r] + bias[o];
        v = v > 0.f ? v : 0.f;
        float* dst = out + ((size_t)b * CO + o) * NN + n;
        asm volatile("global_atomic_max_num_f32 %0, %1, off scope:SCOPE_DEV"
                     :: "v"(dst), "v"(v) : "memory");
      }
    }
  }
}

extern "C" void kernel_launch(void* const* d_in, const int* in_sizes, int n_in,
                              void* d_out, int out_size, void* d_ws, size_t ws_size,
                              hipStream_t stream) {
  const float*     x    = (const float*)d_in[0];
  const long long* eidx = (const long long*)d_in[1];
  const float*     W    = (const float*)d_in[2];
  const float*     bias = (const float*)d_in[3];
  float* out = (float*)d_out;

  _Float16* xT = (_Float16*)d_ws;                                      // 9,633,792 B
  _Float16* Af = (_Float16*)((char*)d_ws + (size_t)BB * NN * CC * 2);  // +294,912 B

  // relu output >= 0, so max-init with 0 is exact (and replay-idempotent).
  (void)hipMemsetAsync(d_out, 0, (size_t)out_size * sizeof(float), stream);

  xt_kernel<<<(BB * CC * NN + 255) / 256, 256, 0, stream>>>(x, xT);
  wprep_kernel<<<(24 * 12 * 32 * 16 + 255) / 256, 256, 0, stream>>>(W, Af);

  dim3 grid(196, 8);  // colblocks (16 nodes each), batch
  edgeconv_kernel<<<grid, 384, 0, stream>>>(xT, Af, eidx, bias, out);
}
